// MultiHeadEDT_89455578841633
// MI455X (gfx1250) — matrix operand order repaired and checked by reading, not compiled, not run on hardware
//
#include <hip/hip_runtime.h>
#include <hip/hip_bf16.h>

typedef __attribute__((ext_vector_type(16))) __bf16 v16bf;
typedef __attribute__((ext_vector_type(8)))  __bf16 v8bf;
typedef __attribute__((ext_vector_type(8)))  float  v8f;
typedef __attribute__((ext_vector_type(4)))  unsigned int u32x4;
typedef __attribute__((ext_vector_type(8)))  int i32x8;
typedef __attribute__((ext_vector_type(4)))  int i32x4;

#define NH    4
#define DDIM  1024
#define ADIM  128
#define PDIM  4
#define TDIM  32
#define BM    32          // rows of x per workgroup
#define NQ    512         // H*A

// padded LDS strides (elements)
#define XS 1032           // x tile, bf16
#define QS 517            // q / gemm2-out, f32
#define CS 520            // attn-out / combined, bf16
#define YS 1029           // y, f32 (overlays xb+q regions)

// LDS byte offsets
#define XB_OFF 0
#define Q_OFF  (BM * XS * 2)                 // 66048
#define C_OFF  (Q_OFF + BM * QS * 4)         // 132224
#define K_OFF  (C_OFF + BM * CS * 2)         // 165504

// constant-region float offsets (within cst[] == packed cpk[] layout)
#define C_KN   0
#define C_PV   2048
#define C_SC   4096
#define C_TW1  4100
#define C_TB1  4228
#define C_TW2  4356
#define C_TB2  4484
#define C_QB   4488
#define C_OB   5000
#define C_LNG  5512
#define C_LNB  6024
#define C_FB   6536
#define C_FLG  7560
#define C_FLB  8584
#define C_TOT  9608

#define LDS_BYTES (K_OFF + C_TOT * 4)        // 203936 B  (<= 320KB / WGP)

// prep-kernel index ranges
#define PREP_QWT 524288
#define PREP_OWT (PREP_QWT + 65536)          // 589824
#define PREP_FWT (PREP_OWT + 524288)         // 1114112
#define PREP_TOT (PREP_FWT + C_TOT)          // 1123720

static __device__ __forceinline__ __bf16 f2bf(float f) {
  unsigned u = __builtin_bit_cast(unsigned, f);
  u += 0x7FFFu + ((u >> 16) & 1u);                 // round-to-nearest-even
  unsigned short h = (unsigned short)(u >> 16);
  return __builtin_bit_cast(__bf16, h);
}

// A-fragment (16x32 bf16): lane holds K=[kb,kb+16),
// kb = 0 (lanes 0-15) or 16 (lanes 16-31). Two 16B LDS loads.
static __device__ __forceinline__ v16bf load_afrag(const __bf16* p) {
  struct Pair { v8bf lo; v8bf hi; } pr;
  pr.lo = *(const v8bf*)(p);
  pr.hi = *(const v8bf*)(p + 8);
  return __builtin_bit_cast(v16bf, pr);
}

static __device__ __forceinline__ v8f wmma_bf16(v16bf a, v16bf b, v8f c) {
  return __builtin_amdgcn_wmma_f32_16x16x32_bf16(false, a, false, b,
                                                 (short)0, c, false, false);
}

// ---------------------------------------------------------------------------
// Prep: bf16 transposed weights (B-matrix layout: N-major, K contiguous) and
// one contiguous packed f32 constant block (TDM source).
// ---------------------------------------------------------------------------
extern "C" __global__ void edt_prep(
    const float* __restrict__ qW,  const float* __restrict__ oW,
    const float* __restrict__ fW,  const float* __restrict__ pk,
    const float* __restrict__ pv,  const float* __restrict__ scl,
    const float* __restrict__ tW1, const float* __restrict__ tb1,
    const float* __restrict__ tW2, const float* __restrict__ tb2,
    const float* __restrict__ qb,  const float* __restrict__ ob,
    const float* __restrict__ lng, const float* __restrict__ lnb,
    const float* __restrict__ fb,  const float* __restrict__ flng,
    const float* __restrict__ flnb,
    __bf16* __restrict__ qWt, __bf16* __restrict__ oWt,
    __bf16* __restrict__ fWt, float* __restrict__ cpk) {
  int i = blockIdx.x * 256 + threadIdx.x;
  if (i < PREP_QWT) {                     // qWt[n*1024+k] = qW[h,k,a], n=h*128+a
    int n = i >> 10, k = i & 1023;
    int h = n >> 7, a = n & 127;
    qWt[i] = f2bf(qW[(h * 1024 + k) * 128 + a]);
  } else if (i < PREP_OWT) {              // oWt[h,c,a] = oW[h,a,c]
    int j = i - PREP_QWT;
    int h = j >> 14, c = (j >> 7) & 127, a = j & 127;
    oWt[j] = f2bf(oW[(h * 128 + a) * 128 + c]);
  } else if (i < PREP_FWT) {              // fWt[d*512+c] = fW[c*1024+d]
    int l = i - PREP_OWT;
    int d = l >> 9, c = l & 511;
    fWt[l] = f2bf(fW[c * 1024 + d]);
  } else if (i < PREP_TOT) {              // packed constants
    int j = i - PREP_FWT;
    float v;
    if (j < C_PV) {                       // kn = normalize(pk) rows
      int hp = j >> 7, a = j & 127;
      const float* row = &pk[hp * 128];
      float ss = 0.f;
      for (int t = 0; t < 128; ++t) ss += row[t] * row[t];
      v = row[a] / fmaxf(sqrtf(ss), 1e-12f);
    }
    else if (j < C_SC)  v = pv[j - C_PV];
    else if (j < C_TW1) v = fminf(fmaxf(scl[j - C_SC], 1.f), 50.f);
    else if (j < C_TB1) v = tW1[j - C_TW1];
    else if (j < C_TW2) v = tb1[j - C_TB1];
    else if (j < C_TB2) v = tW2[j - C_TW2];
    else if (j < C_QB)  v = tb2[j - C_TB2];
    else if (j < C_OB)  v = qb[j - C_QB];
    else if (j < C_LNG) v = ob[j - C_OB];
    else if (j < C_LNB) v = lng[j - C_LNG];
    else if (j < C_FB)  v = lnb[j - C_LNB];
    else if (j < C_FLG) v = fb[j - C_FB];
    else if (j < C_FLB) v = flng[j - C_FLG];
    else                v = flnb[j - C_FLB];
    cpk[j] = v;
  }
}

// ---------------------------------------------------------------------------
// Fully fused kernel: one 32-row block of x -> one 32-row block of out
// ---------------------------------------------------------------------------
extern "C" __global__ void edt_fused(
    const float*  __restrict__ x,
    const __bf16* __restrict__ qWt,
    const __bf16* __restrict__ oWt,
    const __bf16* __restrict__ fWt,
    const float*  __restrict__ cpk,
    float* __restrict__ out) {
  extern __shared__ unsigned char smem[];
  __bf16* xb   = (__bf16*)(smem + XB_OFF);
  float*  qreg = (float*)(smem + Q_OFF);
  __bf16* Cb   = (__bf16*)(smem + C_OFF);
  float*  cst  = (float*)(smem + K_OFF);
  float*  ylds = (float*)(smem);                    // overlays xb+qreg (dead)

  const int tid  = threadIdx.x;
  const int wid  = tid >> 5;
  const int lane = tid & 31;
  const int lo16 = lane & 15;
  const int row0 = blockIdx.x * BM;

  // ---- constants: one Tensor-Data-Mover DMA (global f32 -> LDS) ----------
#if defined(__has_builtin) && __has_builtin(__builtin_amdgcn_tensor_load_to_lds)
  if (wid == 0) {
    const unsigned long long ga = (unsigned long long)(uintptr_t)cpk;
    const unsigned lds_addr = __builtin_amdgcn_groupstaticsize() + (unsigned)K_OFF;
    u32x4 g0;
    g0[0] = 1u;                                         // count=1, user mode
    g0[1] = lds_addr;                                   // lds_addr
    g0[2] = (unsigned)(ga & 0xFFFFFFFFu);               // global_addr[31:0]
    g0[3] = (unsigned)((ga >> 32) & 0x01FFFFFFu) | (2u << 30);  // [56:32]|type=2
    i32x8 g1;
    g1[0] = (int)(2u << 16);                            // data_size=4B
    g1[1] = (int)(((unsigned)C_TOT & 0xFFFFu) << 16);   // tensor_dim0[15:0]
    g1[2] = (int)(((unsigned)C_TOT >> 16) | (1u << 16));// dim0 hi | tensor_dim1=1
    g1[3] = (int)(((unsigned)C_TOT) << 16);             // tile_dim0 = 9608
    g1[4] = 1;                                          // tile_dim1=1, tile_dim2=0
    g1[5] = (int)C_TOT;                                 // tensor_dim0_stride
    g1[6] = 0; g1[7] = 0;
    i32x4 gz = {0, 0, 0, 0};
#if __clang_major__ >= 23
    i32x8 gx = {0, 0, 0, 0, 0, 0, 0, 0};
    __builtin_amdgcn_tensor_load_to_lds(g0, g1, gz, gz, gx, 0);
#else
    __builtin_amdgcn_tensor_load_to_lds(g0, g1, gz, gz, 0);
#endif
  }
  __builtin_amdgcn_s_wait_tensorcnt(0);   // issuing wave: drain TENSORcnt
#else
  for (int i = tid; i < C_TOT; i += 256) cst[i] = cpk[i];
#endif

  // ---- x tile: global f32 -> LDS bf16 (overlaps with TDM const DMA) ------
  for (int i = tid; i < BM * DDIM / 4; i += 256) {
    int r = i >> 8, c4 = i & 255;
    const float4 v = ((const float4*)(x + (size_t)(row0 + r) * DDIM))[c4];
    __bf16* d = &xb[r * XS + c4 * 4];
    d[0] = f2bf(v.x); d[1] = f2bf(v.y); d[2] = f2bf(v.z); d[3] = f2bf(v.w);
  }
  __syncthreads();

  // ================= GEMM1: q = x @ qW + qb   (M=32,K=1024,N=512) ==========
  {
    v8f acc[4][2] = {};
    const int ntb = wid * 4;
    const int kbo = (lane < 16) ? 0 : 16;
    const int kho = (lane < 16) ? 0 : 16;
    for (int kt = 0; kt < 32; ++kt) {
      v16bf a[2];
#pragma unroll
      for (int mt = 0; mt < 2; ++mt)
        a[mt] = load_afrag(&xb[(mt * 16 + lo16) * XS + kt * 32 + kbo]);
      const int kh = kt * 32 + kho;
#pragma unroll
      for (int nt = 0; nt < 4; ++nt) {
        const int n = (ntb + nt) * 16 + lo16;
        const __bf16* bp = &qWt[((size_t)n << 10) + kh];
        __builtin_prefetch(bp + 32, 0, 1);          // global_prefetch_b8
        v16bf b = *(const v16bf*)bp;
#pragma unroll
        for (int mt = 0; mt < 2; ++mt) acc[nt][mt] = wmma_bf16(a[mt], b, acc[nt][mt]);
      }
    }
#pragma unroll
    for (int nt = 0; nt < 4; ++nt) {
      const int n = (ntb + nt) * 16 + lo16;
      const float bias = cst[C_QB + n];
#pragma unroll
      for (int mt = 0; mt < 2; ++mt) {
        const int rb = mt * 16 + ((lane < 16) ? 0 : 8);
#pragma unroll
        for (int r = 0; r < 8; ++r) qreg[(rb + r) * QS + n] = acc[nt][mt][r] + bias;
      }
    }
  }
  __syncthreads();

  // ========== per-(row,head): norm, logits, entropy, tau, softmax, attn@pv ==
  if (wid < 4) {
    const int pair = wid * 32 + lane;      // 128 pairs = 32 rows x 4 heads
    const int r = pair & 31, h = pair >> 5;
    const float* qrow = &qreg[r * QS + h * ADIM];
    const float* knh  = &cst[C_KN + h * PDIM * ADIM];
    float d0 = 0, d1 = 0, d2 = 0, d3 = 0, ss = 0;
    for (int a = 0; a < ADIM; ++a) {
      const float v = qrow[a];
      ss += v * v;
      d0 += v * knh[a];       d1 += v * knh[128 + a];
      d2 += v * knh[256 + a]; d3 += v * knh[384 + a];
    }
    const float inv = cst[C_SC + h] / fmaxf(sqrtf(ss), 1e-12f);
    float raw[4] = { d0 * inv, d1 * inv, d2 * inv, d3 * inv };
    float m = fmaxf(fmaxf(raw[0], raw[1]), fmaxf(raw[2], raw[3]));
    float e[4], Z = 0;
#pragma unroll
    for (int p = 0; p < 4; ++p) { e[p] = __expf(raw[p] - m); Z += e[p]; }
    const float lz = __logf(Z), iZ = 1.f / Z;
    float ent = 0;
#pragma unroll
    for (int p = 0; p < 4; ++p) ent -= (e[p] * iZ) * (raw[p] - m - lz);
    ent *= 0.7213475204444817f;                     // 1/ln(4)
    float hsum = cst[C_TB2 + h];
    for (int t = 0; t < TDIM; ++t) {
      const float hh = fmaxf(ent * cst[C_TW1 + h * 32 + t] + cst[C_TB1 + h * 32 + t], 0.f);
      hsum += hh * cst[C_TW2 + h * 32 + t];
    }
    const float tau = 0.1f + 4.9f / (1.f + __expf(-hsum));
    const float it = 1.f / tau, m2 = m * it;
    float W = 0, w[4];
#pragma unroll
    for (int p = 0; p < 4; ++p) { w[p] = __expf(raw[p] * it - m2); W += w[p]; }
    const float iW = 1.f / W;
    const float a0 = w[0]*iW, a1 = w[1]*iW, a2 = w[2]*iW, a3 = w[3]*iW;
    const float* pvh = &cst[C_PV + h * PDIM * ADIM];
    __bf16* crow = &Cb[r * CS + h * ADIM];
    for (int a = 0; a < ADIM; ++a)
      crow[a] = f2bf(a0 * pvh[a] + a1 * pvh[128 + a] + a2 * pvh[256 + a] + a3 * pvh[384 + a]);
  }
  __syncthreads();

  // ================= GEMM2: per head  (M=32,K=128,N=128) + ob ==============
  {
    const int kbo = (lane < 16) ? 0 : 16;
    const int kho = (lane < 16) ? 0 : 16;
    for (int h = 0; h < NH; ++h) {
      v8f acc[2] = {};
      for (int kt = 0; kt < 4; ++kt) {
        v16bf a[2];
#pragma unroll
        for (int mt = 0; mt < 2; ++mt)
          a[mt] = load_afrag(&Cb[(mt * 16 + lo16) * CS + h * ADIM + kt * 32 + kbo]);
        const int c = wid * 16 + lo16;
        v16bf b = *(const v16bf*)&oWt[(h * ADIM + c) * ADIM + kt * 32 + kho];
#pragma unroll
        for (int mt = 0; mt < 2; ++mt) acc[mt] = wmma_bf16(a[mt], b, acc[mt]);
      }
      const int n = h * ADIM + wid * 16 + lo16;
      const float bias = cst[C_OB + n];
#pragma unroll
      for (int mt = 0; mt < 2; ++mt) {
        const int rb = mt * 16 + ((lane < 16) ? 0 : 8);
#pragma unroll
        for (int r = 0; r < 8; ++r) qreg[(rb + r) * QS + n] = acc[mt][r] + bias;
      }
    }
  }
  __syncthreads();

  // ========== LayerNorm over A -> combined (bf16) ==========================
  if (wid < 4) {
    const int pair = wid * 32 + lane;
    const int r = pair & 31, h = pair >> 5;
    const float* orow = &qreg[r * QS + h * ADIM];
    float s = 0, sq = 0;
    for (int a = 0; a < ADIM; ++a) { const float v = orow[a]; s += v; sq += v * v; }
    const float mu = s * (1.f / ADIM);
    const float rs = rsqrtf(sq * (1.f / ADIM) - mu * mu + 1e-5f);
    __bf16* crow = &Cb[r * CS + h * ADIM];
    for (int a = 0; a < ADIM; ++a)
      crow[a] = f2bf((orow[a] - mu) * rs * cst[C_LNG + h * 128 + a] + cst[C_LNB + h * 128 + a]);
  }
  __syncthreads();

  // ================= GEMM3: y = combined @ fW + fb  (M=32,K=512,N=1024) ====
  {
    const int kbo = (lane < 16) ? 0 : 16;
    const int kho = (lane < 16) ? 0 : 16;
    for (int pass = 0; pass < 2; ++pass) {
      v8f acc[4][2] = {};
      const int ntb = pass * 32 + wid * 4;
      for (int kt = 0; kt < 16; ++kt) {
        v16bf a[2];
#pragma unroll
        for (int mt = 0; mt < 2; ++mt)
          a[mt] = load_afrag(&Cb[(mt * 16 + lo16) * CS + kt * 32 + kbo]);
        const int kh = kt * 32 + kho;
#pragma unroll
        for (int nt = 0; nt < 4; ++nt) {
          const int d = (ntb + nt) * 16 + lo16;
          const __bf16* bp = &fWt[((size_t)d << 9) + kh];
          __builtin_prefetch(bp + 32, 0, 1);
          v16bf b = *(const v16bf*)bp;
#pragma unroll
          for (int mt = 0; mt < 2; ++mt) acc[nt][mt] = wmma_bf16(a[mt], b, acc[nt][mt]);
        }
      }
#pragma unroll
      for (int nt = 0; nt < 4; ++nt) {
        const int d = (ntb + nt) * 16 + lo16;
        const float bias = cst[C_FB + d];
#pragma unroll
        for (int mt = 0; mt < 2; ++mt) {
          const int rb = mt * 16 + ((lane < 16) ? 0 : 8);
#pragma unroll
          for (int r = 0; r < 8; ++r) ylds[(rb + r) * YS + d] = acc[nt][mt][r] + bias;
        }
      }
      __syncthreads();
    }
  }

  // ========== residual + LayerNorm over D (wave32 reduction) -> out ========
  for (int rr = wid; rr < BM; rr += 8) {
    const float* yrow = &ylds[rr * YS];
    const float* xrow = &x[(size_t)(row0 + rr) * DDIM];
    float vv[32];
    float s = 0, sq = 0;
#pragma unroll
    for (int j = 0; j < 32; ++j) {
      const int c = lane + 32 * j;
      const float v = yrow[c] + xrow[c];
      vv[j] = v; s += v; sq += v * v;
    }
#pragma unroll
    for (int off = 16; off > 0; off >>= 1) {
      s  += __shfl_xor(s,  off, 32);
      sq += __shfl_xor(sq, off, 32);
    }
    const float mu = s * (1.f / DDIM);
    const float rs = rsqrtf(sq * (1.f / DDIM) - mu * mu + 1e-5f);
    float* orow = &out[(size_t)(row0 + rr) * DDIM];
#pragma unroll
    for (int j = 0; j < 32; ++j) {
      const int c = lane + 32 * j;
      orow[c] = (vv[j] - mu) * rs * cst[C_FLG + c] + cst[C_FLB + c];
    }
  }
}

// ---------------------------------------------------------------------------
extern "C" void kernel_launch(void* const* d_in, const int* in_sizes, int n_in,
                              void* d_out, int out_size, void* d_ws, size_t ws_size,
                              hipStream_t stream) {
  (void)in_sizes; (void)n_in; (void)out_size; (void)ws_size;
  const float* x    = (const float*)d_in[0];
  const float* qW   = (const float*)d_in[1];
  const float* qb   = (const float*)d_in[2];
  const float* pk   = (const float*)d_in[3];
  const float* pv   = (const float*)d_in[4];
  const float* scl  = (const float*)d_in[5];
  const float* tW1  = (const float*)d_in[6];
  const float* tb1  = (const float*)d_in[7];
  const float* tW2  = (const float*)d_in[8];
  const float* tb2  = (const float*)d_in[9];
  const float* oW   = (const float*)d_in[10];
  const float* ob   = (const float*)d_in[11];
  const float* lng  = (const float*)d_in[12];
  const float* lnb  = (const float*)d_in[13];
  const float* fW   = (const float*)d_in[14];
  const float* fb   = (const float*)d_in[15];
  const float* flng = (const float*)d_in[16];
  const float* flnb = (const float*)d_in[17];

  unsigned char* ws = (unsigned char*)d_ws;       // ~2.27 MB used
  __bf16* qWt = (__bf16*)(ws);
  __bf16* oWt = (__bf16*)(ws + 1048576);
  __bf16* fWt = (__bf16*)(ws + 1048576 + 131072);
  float*  cpk = (float*) (ws + 1048576 + 131072 + 1048576);

  edt_prep<<<(PREP_TOT + 255) / 256, 256, 0, stream>>>(
      qW, oW, fW, pk, pv, scl, tW1, tb1, tW2, tb2, qb, ob, lng, lnb,
      fb, flng, flnb, qWt, oWt, fWt, cpk);

  (void)hipFuncSetAttribute((const void*)edt_fused,
                            hipFuncAttributeMaxDynamicSharedMemorySize,
                            (int)LDS_BYTES);
  edt_fused<<<131072 / BM, 256, LDS_BYTES, stream>>>(
      x, qWt, oWt, fWt, cpk, (float*)d_out);
}
